// SpatialSelfAttention_35175782154615
// MI455X (gfx1250) — compile-verified, matrix operand order checked by reading, not run on hardware
//
#include <hip/hip_runtime.h>

// ---------------- types ----------------
typedef __attribute__((ext_vector_type(16))) __bf16 v16bf;
typedef __attribute__((ext_vector_type(8)))  __bf16 v8bf;
typedef __attribute__((ext_vector_type(8)))  float  v8f;
typedef __attribute__((ext_vector_type(4)))  float  f32x4;
typedef __attribute__((ext_vector_type(4)))  int    i32x4;

#define WMMA_BF16(A,B,C) __builtin_amdgcn_wmma_f32_16x16x32_bf16(false,(A),false,(B),(short)0,(C),false,false)

static __device__ __forceinline__ v16bf cat8(v8bf lo, v8bf hi) {
  return __builtin_shufflevector(lo, hi, 0,1,2,3,4,5,6,7,8,9,10,11,12,13,14,15);
}

// ---- CDNA5 async global->LDS copy (ASYNCcnt), guarded with sync fallback ----
#if __has_builtin(__builtin_amdgcn_global_load_async_to_lds_b128) && \
    __has_builtin(__builtin_amdgcn_s_wait_asynccnt)
#define HAVE_ASYNC 1
#else
#define HAVE_ASYNC 0
#endif

#if HAVE_ASYNC
typedef __attribute__((address_space(1))) i32x4* gas_p;   // pointer to int4 in global AS
typedef __attribute__((address_space(3))) i32x4* las_p;   // pointer to int4 in LDS AS
#endif

static __device__ __forceinline__ void cp16(const void* g, void* l) {
#if HAVE_ASYNC
  __builtin_amdgcn_global_load_async_to_lds_b128((gas_p)g, (las_p)l, 0, 0);
#else
  *(v8bf*)l = *(const v8bf*)g;
#endif
}
static __device__ __forceinline__ void cp_wait() {
#if HAVE_ASYNC
  __builtin_amdgcn_s_wait_asynccnt(0);
#endif
}

static constexpr int B_  = 8;
static constexpr int C_  = 512;
static constexpr int N_  = 4096;   // H*W
static constexpr int G_  = 32;
static constexpr int CPG = 16;     // channels per group

// ============================================================
// Kernel 1: GroupNorm statistics. 1 block per (b,g), 256 thr.
// ============================================================
__global__ __attribute__((amdgpu_flat_work_group_size(256, 256)))
void k_gn_stats(const float* __restrict__ x, float* __restrict__ stats) {
  int bg = blockIdx.x;
  int b = bg >> 5, g = bg & 31;
  const float* xp = x + ((size_t)b * C_ + (size_t)g * CPG) * N_;
  float s = 0.f, s2 = 0.f;
  for (int i = threadIdx.x; i < CPG * N_; i += blockDim.x) {
    float v = xp[i];
    s += v; s2 += v * v;
  }
  for (int o = 16; o >= 1; o >>= 1) {
    s  += __shfl_xor(s,  o, 32);
    s2 += __shfl_xor(s2, o, 32);
  }
  __shared__ float rs[8], rs2[8];
  int wid = threadIdx.x >> 5, lane = threadIdx.x & 31;
  if (lane == 0) { rs[wid] = s; rs2[wid] = s2; }
  __syncthreads();
  if (threadIdx.x == 0) {
    float t = 0.f, t2 = 0.f;
    for (int i = 0; i < 8; i++) { t += rs[i]; t2 += rs2[i]; }
    float inv = 1.0f / (float)(CPG * N_);
    float mu = t * inv;
    float var = t2 * inv - mu * mu;
    stats[bg * 2 + 0] = mu;
    stats[bg * 2 + 1] = rsqrtf(var + 1e-6f);
  }
}

// ============================================================
// Kernel 2: h = gamma*(x-mu)*rstd + beta, transposed to bf16 [B,N,C].
// grid (ntile=64, ctile=8, b=8), block 256. LDS tile transpose.
// ============================================================
__global__ __attribute__((amdgpu_flat_work_group_size(256, 256)))
void k_build_h(const float* __restrict__ x, const float* __restrict__ gw,
               const float* __restrict__ gb, const float* __restrict__ stats,
               __bf16* __restrict__ hb) {
  __shared__ float tile[64][65];
  int n0 = blockIdx.x * 64, c0 = blockIdx.y * 64, b = blockIdx.z;
  const float* xp = x + ((size_t)b * C_ + c0) * N_ + n0;
  for (int i = threadIdx.x; i < 64 * 64; i += 256) {
    int cl = i >> 6, nl = i & 63;
    int c = c0 + cl, g = c >> 4;
    float mu   = stats[(b * G_ + g) * 2 + 0];
    float rstd = stats[(b * G_ + g) * 2 + 1];
    float v = xp[(size_t)cl * N_ + nl];
    tile[cl][nl] = (v - mu) * rstd * gw[c] + gb[c];
  }
  __syncthreads();
  __bf16* hp = hb + ((size_t)b * N_ + n0) * C_ + c0;
  for (int i = threadIdx.x; i < 64 * 64; i += 256) {
    int nl = i >> 6, cl = i & 63;
    hp[(size_t)nl * C_ + cl] = (__bf16)tile[cl][nl];
  }
}

// ============================================================
// Kernel 3: QKV projection GEMM (WMMA bf16).
//   Q,K -> [B,N,C] bf16 ; V -> [B,C,N] bf16 (channel-major for PV B-frags)
// grid (ntile=64, otile=8, b=8), block 128 (4 waves). Tile 64n x 64o, K-chunk 32.
// ============================================================
__global__ __attribute__((amdgpu_flat_work_group_size(128, 128), amdgpu_waves_per_eu(2)))
void k_qkv(const __bf16* __restrict__ hb,
           const float* __restrict__ wq, const float* __restrict__ bq,
           const float* __restrict__ wk, const float* __restrict__ bk,
           const float* __restrict__ wv, const float* __restrict__ bv,
           __bf16* __restrict__ qo, __bf16* __restrict__ ko, __bf16* __restrict__ vo) {
  __shared__ __bf16 Ah[64 * 40];
  __shared__ __bf16 Wq[64 * 40];
  __shared__ __bf16 Wk[64 * 40];
  __shared__ __bf16 Wv[64 * 40];
  int n0 = blockIdx.x * 64, o0 = blockIdx.y * 64, b = blockIdx.z;
  int lane = threadIdx.x & 31, w = threadIdx.x >> 5;
  int mr = lane & 15;
  int off1 = (lane < 16) ? 0 : 8;

  v8f aq[4] = {}, ak[4] = {}, av[4] = {};

  for (int kc = 0; kc < C_; kc += 32) {
    const __bf16* ag = hb + ((size_t)b * N_ + n0) * C_ + kc;
    for (int i = threadIdx.x; i < 256; i += 128) {
      int r = i >> 2, cc = (i & 3) * 8;
      cp16(ag + (size_t)r * C_ + cc, &Ah[r * 40 + cc]);
    }
    for (int i = threadIdx.x; i < 2048; i += 128) {
      int r = i >> 5, cc = i & 31;
      size_t gi = (size_t)(o0 + r) * C_ + kc + cc;
      Wq[r * 40 + cc] = (__bf16)wq[gi];
      Wk[r * 40 + cc] = (__bf16)wk[gi];
      Wv[r * 40 + cc] = (__bf16)wv[gi];
    }
    cp_wait();
    __syncthreads();
    int arow = w * 16 + mr;
    v16bf a = cat8(*(const v8bf*)&Ah[arow * 40 + off1],
                   *(const v8bf*)&Ah[arow * 40 + 16 + off1]);
#pragma unroll
    for (int f = 0; f < 4; f++) {
      int orow = f * 16 + mr;
      v16bf bfq = cat8(*(const v8bf*)&Wq[orow * 40 + off1], *(const v8bf*)&Wq[orow * 40 + 16 + off1]);
      v16bf bfk = cat8(*(const v8bf*)&Wk[orow * 40 + off1], *(const v8bf*)&Wk[orow * 40 + 16 + off1]);
      v16bf bfv = cat8(*(const v8bf*)&Wv[orow * 40 + off1], *(const v8bf*)&Wv[orow * 40 + 16 + off1]);
      aq[f] = WMMA_BF16(a, bfq, aq[f]);
      ak[f] = WMMA_BF16(a, bfk, ak[f]);
      av[f] = WMMA_BF16(a, bfv, av[f]);
    }
    __syncthreads();
  }

  // Epilogue: bias + stores.
  int rb = (lane >> 4) << 3;          // 0 or 8 (row offset of this lane half)
  int nbase = n0 + w * 16;
#pragma unroll
  for (int f = 0; f < 4; f++) {
    int oc = o0 + f * 16 + mr;
    float biasq = bq[oc], biask = bk[oc], biasv = bv[oc];
#pragma unroll
    for (int t = 0; t < 8; t++) {
      size_t a0 = ((size_t)b * N_ + nbase + rb + t) * C_ + oc;
      qo[a0] = (__bf16)(aq[f][t] + biasq);
      ko[a0] = (__bf16)(ak[f][t] + biask);
    }
    v8bf pv;
#pragma unroll
    for (int t = 0; t < 8; t++) pv[t] = (__bf16)(av[f][t] + biasv);
    *(v8bf*)&vo[((size_t)b * C_ + oc) * N_ + nbase + rb] = pv;
  }
}

// ============================================================
// Kernel 4: fused flash attention (WMMA bf16, online softmax).
// grid (itile=64, b=8), block 128 (4 waves); wave owns 16 i-rows x 512 channels.
// Dynamic LDS: Qt[64][520] + Kt[64][520] + Vt[512][72] + P[4][16][72] = 216064 B
// ============================================================
__global__ __attribute__((amdgpu_flat_work_group_size(128, 128), amdgpu_waves_per_eu(2)))
void k_attn(const __bf16* __restrict__ q, const __bf16* __restrict__ k,
            const __bf16* __restrict__ v, __bf16* __restrict__ ao) {
  extern __shared__ char smem[];
  __bf16* Qt = (__bf16*)smem;                                        // [64][520]
  __bf16* Kt = (__bf16*)(smem + 64 * 520 * 2);                       // [64][520]
  __bf16* Vt = (__bf16*)(smem + 2 * 64 * 520 * 2);                   // [512][72]
  __bf16* Pl = (__bf16*)(smem + 2 * 64 * 520 * 2 + 512 * 72 * 2);    // [4][16][72]

  int b = blockIdx.y;
  int lane = threadIdx.x & 31, w = threadIdx.x >> 5;
  int mr = lane & 15;
  int off1 = (lane < 16) ? 0 : 8;
  int rb = (lane >> 4) << 3;
  const float scale = 0.044194173824159216f;  // 512^-0.5

  // stage Q tile (this block's 64 query rows) once
  {
    const __bf16* qg = q + ((size_t)b * N_ + blockIdx.x * 64) * C_;
    for (int i = threadIdx.x; i < 64 * 64; i += 128) {
      int r = i >> 6, cc = (i & 63) * 8;
      cp16(qg + (size_t)r * C_ + cc, &Qt[r * 520 + cc]);
    }
  }

  v8f oacc[32] = {};
  float mrun[8], lrun[8];
#pragma unroll
  for (int t = 0; t < 8; t++) { mrun[t] = -3.0e38f; lrun[t] = 0.f; }

  const __bf16* kbase = k + (size_t)b * N_ * C_;
  const __bf16* vbase = v + (size_t)b * C_ * N_;
  const int arow = w * 16 + mr;

  for (int jt = 0; jt < N_; jt += 64) {
    // ---- stage K tile [j][c] and V^T tile [c][j] (async -> LDS) ----
    const __bf16* kg = kbase + (size_t)jt * C_;
    for (int i = threadIdx.x; i < 64 * 64; i += 128) {
      int r = i >> 6, cc = (i & 63) * 8;
      cp16(kg + (size_t)r * C_ + cc, &Kt[r * 520 + cc]);
    }
    const __bf16* vg = vbase + jt;
    for (int i = threadIdx.x; i < 512 * 8; i += 128) {
      int r = i >> 3, cc = (i & 7) * 8;
      cp16(vg + (size_t)r * N_ + cc, &Vt[r * 72 + cc]);
    }
    if (jt + 64 < N_) {  // prefetch next tiles into L2 (global_prefetch_b8)
      __builtin_prefetch(kg + (size_t)64 * C_ + threadIdx.x * 16, 0, 1);
      __builtin_prefetch(vg + 64 + (size_t)threadIdx.x * 4 * N_, 0, 1);
    }
    cp_wait();
    __syncthreads();

    // ---- S = (Q K^T) * scale : 64 WMMAs ----
    v8f s[4] = {};
#pragma unroll
    for (int kc = 0; kc < 16; kc++) {
      const __bf16* qr = &Qt[arow * 520 + kc * 32 + off1];
      v16bf a = cat8(*(const v8bf*)qr, *(const v8bf*)(qr + 16));
#pragma unroll
      for (int f = 0; f < 4; f++) {
        const __bf16* kb = &Kt[(f * 16 + mr) * 520 + kc * 32 + off1];
        v16bf bb = cat8(*(const v8bf*)kb, *(const v8bf*)(kb + 16));
        s[f] = WMMA_BF16(a, bb, s[f]);
      }
    }
#pragma unroll
    for (int f = 0; f < 4; f++)
#pragma unroll
      for (int t = 0; t < 8; t++) s[f][t] *= scale;

    // ---- online softmax (rows live per 16-lane half; reduce with xor<=8) ----
    float corr[8];
#pragma unroll
    for (int t = 0; t < 8; t++) {
      float mx = fmaxf(fmaxf(s[0][t], s[1][t]), fmaxf(s[2][t], s[3][t]));
#pragma unroll
      for (int o = 8; o >= 1; o >>= 1) mx = fmaxf(mx, __shfl_xor(mx, o, 32));
      float mnew = fmaxf(mrun[t], mx);
      corr[t] = __expf(mrun[t] - mnew);
      float ps = 0.f;
#pragma unroll
      for (int f = 0; f < 4; f++) {
        float p = __expf(s[f][t] - mnew);
        s[f][t] = p; ps += p;
      }
#pragma unroll
      for (int o = 8; o >= 1; o >>= 1) ps += __shfl_xor(ps, o, 32);
      lrun[t] = lrun[t] * corr[t] + ps;
      mrun[t] = mnew;
    }
#pragma unroll
    for (int cf = 0; cf < 32; cf++)
#pragma unroll
      for (int t = 0; t < 8; t++) oacc[cf][t] *= corr[t];

    // ---- P -> LDS (D-layout -> A-layout bounce), then PV: 64 WMMAs ----
    __bf16* pw = Pl + w * (16 * 72);
#pragma unroll
    for (int f = 0; f < 4; f++)
#pragma unroll
      for (int t = 0; t < 8; t++)
        pw[(rb + t) * 72 + f * 16 + mr] = (__bf16)s[f][t];

#pragma unroll
    for (int jc = 0; jc < 2; jc++) {
      const __bf16* pr = pw + mr * 72 + jc * 32 + off1;
      v16bf pa = cat8(*(const v8bf*)pr, *(const v8bf*)(pr + 16));
#pragma unroll
      for (int cf = 0; cf < 32; cf++) {
        const __bf16* vb = &Vt[(cf * 16 + mr) * 72 + jc * 32 + off1];
        v16bf bb = cat8(*(const v8bf*)vb, *(const v8bf*)(vb + 16));
        oacc[cf] = WMMA_BF16(pa, bb, oacc[cf]);
      }
    }
    __syncthreads();
  }

  // ---- finalize: divide by row sums, store [B,N,C] bf16 ----
#pragma unroll
  for (int t = 0; t < 8; t++) lrun[t] = 1.0f / lrun[t];
  int i0 = blockIdx.x * 64 + w * 16;
#pragma unroll
  for (int cf = 0; cf < 32; cf++) {
    int c = cf * 16 + mr;
#pragma unroll
    for (int t = 0; t < 8; t++)
      ao[((size_t)b * N_ + i0 + rb + t) * C_ + c] = (__bf16)(oacc[cf][t] * lrun[t]);
  }
}

// ============================================================
// Kernel 5: proj GEMM + bias + residual -> f32 [B,C,N].
// grid (ntile=64, otile=8, b=8), block 128 (4 waves).
// ============================================================
__global__ __attribute__((amdgpu_flat_work_group_size(128, 128), amdgpu_waves_per_eu(2)))
void k_proj(const __bf16* __restrict__ ao, const float* __restrict__ wp,
            const float* __restrict__ bp, const float* __restrict__ x,
            float* __restrict__ out) {
  __shared__ __bf16 Ah[64 * 40];
  __shared__ __bf16 Wl[64 * 40];
  int n0 = blockIdx.x * 64, o0 = blockIdx.y * 64, b = blockIdx.z;
  int lane = threadIdx.x & 31, w = threadIdx.x >> 5;
  int mr = lane & 15;
  int off1 = (lane < 16) ? 0 : 8;

  v8f acc[4] = {};
  for (int kc = 0; kc < C_; kc += 32) {
    const __bf16* ag = ao + ((size_t)b * N_ + n0) * C_ + kc;
    for (int i = threadIdx.x; i < 256; i += 128) {
      int r = i >> 2, cc = (i & 3) * 8;
      cp16(ag + (size_t)r * C_ + cc, &Ah[r * 40 + cc]);
    }
    for (int i = threadIdx.x; i < 2048; i += 128) {
      int r = i >> 5, cc = i & 31;
      Wl[r * 40 + cc] = (__bf16)wp[(size_t)(o0 + r) * C_ + kc + cc];
    }
    cp_wait();
    __syncthreads();
    int arow = w * 16 + mr;
    v16bf a = cat8(*(const v8bf*)&Ah[arow * 40 + off1],
                   *(const v8bf*)&Ah[arow * 40 + 16 + off1]);
#pragma unroll
    for (int f = 0; f < 4; f++) {
      const __bf16* wb = &Wl[(f * 16 + mr) * 40 + off1];
      v16bf bb = cat8(*(const v8bf*)wb, *(const v8bf*)(wb + 16));
      acc[f] = WMMA_BF16(a, bb, acc[f]);
    }
    __syncthreads();
  }

  int rb = (lane >> 4) << 3;
  int nbase = n0 + w * 16 + rb;
#pragma unroll
  for (int f = 0; f < 4; f++) {
    int oc = o0 + f * 16 + mr;
    float bias = bp[oc];
    size_t base = ((size_t)b * C_ + oc) * N_ + nbase;
    f32x4 x0 = *(const f32x4*)(x + base);
    f32x4 x1 = *(const f32x4*)(x + base + 4);
    f32x4 r0, r1;
#pragma unroll
    for (int t = 0; t < 4; t++) {
      r0[t] = acc[f][t]     + bias + x0[t];
      r1[t] = acc[f][t + 4] + bias + x1[t];
    }
    *(f32x4*)(out + base)     = r0;
    *(f32x4*)(out + base + 4) = r1;
  }
}

// ============================================================
extern "C" void kernel_launch(void* const* d_in, const int* in_sizes, int n_in,
                              void* d_out, int out_size, void* d_ws, size_t ws_size,
                              hipStream_t stream) {
  const float* x    = (const float*)d_in[0];
  const float* gn_w = (const float*)d_in[1];
  const float* gn_b = (const float*)d_in[2];
  const float* wq   = (const float*)d_in[3];
  const float* bq   = (const float*)d_in[4];
  const float* wk   = (const float*)d_in[5];
  const float* bk   = (const float*)d_in[6];
  const float* wv   = (const float*)d_in[7];
  const float* bv   = (const float*)d_in[8];
  const float* wp   = (const float*)d_in[9];
  const float* bp   = (const float*)d_in[10];

  const size_t MAT = (size_t)B_ * N_ * C_ * sizeof(__bf16);  // 32 MB
  if (ws_size < 4096 + 4 * MAT) return;

  char* ws = (char*)d_ws;
  float*  stats = (float*)ws;
  __bf16* hb = (__bf16*)(ws + 4096);
  __bf16* qb = (__bf16*)(ws + 4096 + 1 * MAT);
  __bf16* kb = (__bf16*)(ws + 4096 + 2 * MAT);
  __bf16* vb = (__bf16*)(ws + 4096 + 3 * MAT);
  __bf16* aob = hb;  // h is dead after k_qkv; alias attention output onto it

  k_gn_stats<<<B_ * G_, 256, 0, stream>>>(x, stats);
  k_build_h<<<dim3(64, 8, B_), 256, 0, stream>>>(x, gn_w, gn_b, stats, hb);
  k_qkv<<<dim3(64, 8, B_), 128, 0, stream>>>(hb, wq, bq, wk, bk, wv, bv, qb, kb, vb);

  const int attn_lds = 2 * 64 * 520 * 2 + 512 * 72 * 2 + 4 * 16 * 72 * 2;  // 216064 B
  (void)hipFuncSetAttribute(reinterpret_cast<const void*>(k_attn),
                            hipFuncAttributeMaxDynamicSharedMemorySize, attn_lds);
  k_attn<<<dim3(64, B_), 128, attn_lds, stream>>>(qb, kb, vb, aob);

  k_proj<<<dim3(64, 8, B_), 128, 0, stream>>>(aob, wp, bp, x, (float*)d_out);
}
